// TradingGraphMambaEncoder_43937515438701
// MI455X (gfx1250) — matrix operand order repaired and checked by reading, not compiled, not run on hardware
//
#include <hip/hip_runtime.h>
#include <math.h>

// ---------------- model constants ----------------
#define BB   4
#define TT   512
#define DD   384
#define SS   32
#define EE   6
#define HH   512
#define RR   5
#define LL   2
#define DIx  768
#define DTRx 24
#define NXx  88      // DTR + 2*S
#define MM   2048    // B*T

typedef __attribute__((ext_vector_type(16))) _Float16 v16h;
typedef __attribute__((ext_vector_type(8)))  float    v8f;
typedef __attribute__((ext_vector_type(8)))  _Float16 h8;

__device__ __forceinline__ float siluf(float v)     { return v / (1.f + __expf(-v)); }
__device__ __forceinline__ float softplusf(float v) { return v > 20.f ? v : log1pf(expf(v)); }

// ---------------------------------------------------------------------------
// WMMA GEMM on pre-converted f16 operands:
//   Cout[M,N](f32) / CoutH[M,N](f16) =
//       act( A16[M,K] @ Wt[N,K]^T + colBias + batchBias + Cin )
// A16 row-major f16 (lda halves); Wt row-major f16 TRANSPOSED weights [N][K].
// 256 threads = 8 waves as 4(M) x 2(N); block tile 64x128; wave tile 16x64
// -> 4 v_wmma_f32_16x16x32_f16 per K-step, A-fragment reused across 4 B-frags.
// All staging is h8 (b128) loads/stores; no conversions in the hot loop.
// Requires M%64==0, K%32==0, lda%8==0. N guarded vs BN=128.
// ---------------------------------------------------------------------------
__global__ __launch_bounds__(256) void tg_gemm_wmma(
    const _Float16* __restrict__ A, const _Float16* __restrict__ Wt,
    const float* __restrict__ colBias, const float* __restrict__ batchBias,
    const float* __restrict__ Cin, float* __restrict__ Cout,
    _Float16* __restrict__ CoutH,
    int M, int N, int K, int lda, int ldc, int act)
{
    __shared__ _Float16 As[64][40];    // [row][k]  stride 80B -> 16B-aligned frags
    __shared__ _Float16 Bs[128][40];   // [col][k]  (weights pre-transposed)

    const int tid   = threadIdx.x;
    const int lane  = tid & 31;
    const int wave  = tid >> 5;
    const int waveM = wave >> 1;       // 0..3
    const int waveN = wave & 1;        // 0..1
    const int bm    = blockIdx.y * 64;
    const int bn    = blockIdx.x * 128;

    // staging: A 64x32 halves = 256 h8 slots; B 128x32 halves = 512 h8 slots
    const int aR = tid >> 2;           // 0..63
    const int aC = (tid & 3) * 8;      // k offset
    const int bCl = tid >> 2;          // col slots t>>2 and +64
    const int bK  = (tid & 3) * 8;
    const int c0  = bn + bCl, c1 = c0 + 64;

    h8 aV, bV0, bV1;
    const h8 hz{};

    auto loadTiles = [&](int k0) {
        aV  = *(const h8*)&A[(size_t)(bm + aR) * lda + k0 + aC];
        bV0 = (c0 < N) ? *(const h8*)&Wt[(size_t)c0 * K + k0 + bK] : hz;
        bV1 = (c1 < N) ? *(const h8*)&Wt[(size_t)c1 * K + k0 + bK] : hz;
    };
    auto storeTiles = [&]() {
        *(h8*)&As[aR][aC]       = aV;
        *(h8*)&Bs[bCl][bK]      = bV0;
        *(h8*)&Bs[bCl + 64][bK] = bV1;
    };

    v8f acc[4] = {v8f{}, v8f{}, v8f{}, v8f{}};

    const int half = lane >> 4;
    const int l15  = lane & 15;
    const int fr   = l15 + waveM * 16;   // A fragment row in LDS
    const int ah   = half * 8;           // A frag K base (ISA 7.12.2 layout)
    const int bh   = half * 8;           // B frag K base
    const int cb   = waveN * 64 + l15;   // B fragment column base

    loadTiles(0);
    storeTiles();
    __syncthreads();

    const int nk = K >> 5;
    for (int kt = 0; kt < nk; ++kt) {
        const bool more = (kt + 1) < nk;
        if (more) loadTiles((kt + 1) << 5);
        if (kt + 2 < nk) {
            __builtin_prefetch(&A[(size_t)(bm + aR) * lda + ((kt + 2) << 5) + aC], 0, 1);
            __builtin_prefetch(&Wt[(size_t)c0 * K + ((kt + 2) << 5) + bK], 0, 1);
        }

        // issue ALL fragment LDS loads first (lets waits be partial dscnt)
        h8 a0 = *(const h8*)&As[fr][ah];
        h8 a1 = *(const h8*)&As[fr][16 + ah];
        h8 b0[4], b1[4];
#pragma unroll
        for (int n = 0; n < 4; ++n) {
            b0[n] = *(const h8*)&Bs[cb + n * 16][bh];
            b1[n] = *(const h8*)&Bs[cb + n * 16][bh + 16];
        }
        v16h af;
#pragma unroll
        for (int j = 0; j < 8; ++j) { af[j] = a0[j]; af[j + 8] = a1[j]; }
#pragma unroll
        for (int n = 0; n < 4; ++n) {
            v16h bf;
#pragma unroll
            for (int j = 0; j < 8; ++j) { bf[j] = b0[n][j]; bf[j + 8] = b1[n][j]; }
            acc[n] = __builtin_amdgcn_wmma_f32_16x16x32_f16(
                         false, af, false, bf, (short)0, acc[n], false, false);
        }
        __syncthreads();
        if (more) { storeTiles(); __syncthreads(); }
    }

    // epilogue: D element v -> row = tile + half*8 + v, col = tile + (lane&15)
    const int row0 = bm + waveM * 16 + half * 8;
#pragma unroll
    for (int n = 0; n < 4; ++n) {
        const int col = bn + waveN * 64 + n * 16 + l15;
        if (col < N) {
#pragma unroll
            for (int v = 0; v < 8; ++v) {
                int row = row0 + v;
                float val = acc[n][v];
                if (colBias)   val += colBias[col];
                if (batchBias) val += batchBias[(row / TT) * N + col];
                if (Cin)       val += Cin[(size_t)row * ldc + col];
                if (act == 1)  val = siluf(val);
                if (Cout)  Cout[(size_t)row * ldc + col] = val;
                if (CoutH) CoutH[(size_t)row * ldc + col] = (_Float16)val;
            }
        }
    }
}

// ---------------- weight convert+transpose: f32 [K][N] -> f16 [N][K] --------
__global__ void tg_w16t(const float* __restrict__ w, _Float16* __restrict__ wt,
                        int K, int N)
{
    int idx = blockIdx.x * blockDim.x + threadIdx.x;
    if (idx >= K * N) return;
    int k = idx / N, n = idx % N;
    wt[(size_t)n * K + k] = (_Float16)w[idx];
}

// ---------------- LayerNorm: one row per block; f32 and/or f16 out ----------
__global__ __launch_bounds__(128) void tg_ln(
    const float* __restrict__ x, const float* __restrict__ g,
    const float* __restrict__ b, float* __restrict__ o32,
    _Float16* __restrict__ o16)
{
    const int row = blockIdx.x, tid = threadIdx.x;
    const float* xr = x + (size_t)row * DD;
    float s = 0.f, ss = 0.f;
    for (int j = tid; j < DD; j += 128) { float v = xr[j]; s += v; ss += v * v; }
    __shared__ float red0[128], red1[128];
    red0[tid] = s; red1[tid] = ss; __syncthreads();
    for (int off = 64; off > 0; off >>= 1) {
        if (tid < off) { red0[tid] += red0[tid + off]; red1[tid] += red1[tid + off]; }
        __syncthreads();
    }
    float mean = red0[0] / DD;
    float var  = red1[0] / DD - mean * mean;
    float rstd = rsqrtf(var + 1e-5f);
    for (int j = tid; j < DD; j += 128) {
        float val = (xr[j] - mean) * rstd * g[j] + b[j];
        if (o32) o32[(size_t)row * DD + j] = val;
        if (o16) o16[(size_t)row * DD + j] = (_Float16)val;
    }
}

// ---------------- FiLM MLP (tiny): gb[b, 2D] -------------------------------
__global__ __launch_bounds__(256) void tg_film(
    const float* __restrict__ regime, const float* __restrict__ w1,
    const float* __restrict__ b1, const float* __restrict__ w2,
    const float* __restrict__ b2, float* __restrict__ gb)
{
    const int b = blockIdx.x, tid = threadIdx.x;
    __shared__ float hid[DD];
    float r[RR];
#pragma unroll
    for (int i = 0; i < RR; ++i) r[i] = regime[b * RR + i];
    for (int j = tid; j < DD; j += 256) {
        float s = b1[j];
#pragma unroll
        for (int i = 0; i < RR; ++i) s += r[i] * w1[i * DD + j];
        hid[j] = siluf(s);
    }
    __syncthreads();
    for (int j = tid; j < 2 * DD; j += 256) {
        float s = b2[j];
        for (int d = 0; d < DD; ++d) s += hid[d] * w2[d * (2 * DD) + j];
        gb[b * (2 * DD) + j] = s;
    }
}

__global__ void tg_film_apply(const float* __restrict__ x,
                              const float* __restrict__ gb,
                              float* __restrict__ h, int total)
{
    int idx = blockIdx.x * blockDim.x + threadIdx.x;
    if (idx >= total) return;
    int d = idx % DD;
    int b = idx / (TT * DD);
    h[idx] = x[idx] * (gb[b * 768 + d] + 1.f) + gb[b * 768 + DD + d];
}

// ---------------- dt = softplus(x_dbl[:, :24] @ dt_w + dt_b) ----------------
__global__ void tg_dt(const float* __restrict__ xdbl, const float* __restrict__ dtw,
                      const float* __restrict__ dtb, float* __restrict__ dt)
{
    int idx = blockIdx.x * blockDim.x + threadIdx.x;
    if (idx >= MM * DIx) return;
    int row = idx / DIx, di = idx % DIx;
    const float* xd = xdbl + (size_t)row * NXx;
    float acc = dtb[di];
#pragma unroll
    for (int k = 0; k < DTRx; ++k) acc += xd[k] * dtw[k * DIx + di];
    dt[idx] = softplusf(acc);
}

// ---------------- complex SSM scan: one wave per (b, di); lane == state s ---
__global__ __launch_bounds__(256) void tg_scan(
    const float* __restrict__ dt, const float* __restrict__ xz,
    const float* __restrict__ xdbl, const float* __restrict__ alr,
    const float* __restrict__ aim, float* __restrict__ y)
{
    const int wave = blockIdx.x * (blockDim.x >> 5) + (threadIdx.x >> 5);
    const int lane = threadIdx.x & 31;
    const int b  = wave / DIx;
    const int di = wave % DIx;
    if (b >= BB) return;

    const float ar = -__expf(alr[di * SS + lane]);
    const float ai = aim[di * SS + lane];
    float hr = 0.f, hi = 0.f;

    for (int t = 0; t < TT; ++t) {
        const size_t row = (size_t)b * TT + t;
        float dtv = dt[row * DIx + di];                 // broadcast within wave
        float xt  = xz[row * 1536 + di];                // x_ssm slice of xz
        float bt  = xdbl[row * NXx + DTRx + lane];      // B_t[s]
        float ct  = xdbl[row * NXx + DTRx + SS + lane]; // C_t[s]

        float e = __expf(ar * dtv);
        float sn, cs; __sincosf(ai * dtv, &sn, &cs);
        float dAr = e * cs, dAi = e * sn;
        float u   = dtv * xt * bt;
        float nhr = dAr * hr - dAi * hi + u;
        float nhi = dAr * hi + dAi * hr;
        hr = nhr; hi = nhi;

        float contrib = hr * ct;
#pragma unroll
        for (int off = 16; off > 0; off >>= 1)
            contrib += __shfl_xor(contrib, off, 32);
        if (lane == 0) y[row * DIx + di] = contrib;
    }
}

// ---------------- y16 = f16( (y + D*x_ssm) * silu(z) ) ----------------------
__global__ void tg_gate(const float* __restrict__ xz, const float* __restrict__ dvec,
                        const float* __restrict__ y, _Float16* __restrict__ y16)
{
    int idx = blockIdx.x * blockDim.x + threadIdx.x;
    if (idx >= MM * DIx) return;
    int row = idx / DIx, di = idx % DIx;
    float z  = xz[(size_t)row * 1536 + DIx + di];
    float xs = xz[(size_t)row * 1536 + di];
    y16[idx] = (_Float16)((y[idx] + dvec[di] * xs) * siluf(z));
}

// ---------------- router regime-bias (per batch) ----------------------------
__global__ void tg_rbias(const float* __restrict__ regime, const float* __restrict__ rw1,
                         const float* __restrict__ rb1, float* __restrict__ rbias)
{
    int b = blockIdx.x, j = threadIdx.x;   // 384 threads
    float s = rb1[j];
#pragma unroll
    for (int r = 0; r < RR; ++r) s += regime[b * RR + r] * rw1[(DD + r) * DD + j];
    rbias[b * DD + j] = s;
}

// ---------------- router logits: gl = g1 @ r_w2 + r_b2 ----------------------
__global__ void tg_gl(const float* __restrict__ g1, const float* __restrict__ rw2,
                      const float* __restrict__ rb2, float* __restrict__ gl)
{
    int idx = blockIdx.x * blockDim.x + threadIdx.x;
    if (idx >= MM * EE) return;
    int row = idx / EE, e = idx % EE;
    const float* gr = g1 + (size_t)row * DD;
    float acc = rb2[e];
    for (int d = 0; d < DD; ++d) acc += gr[d] * rw2[d * EE + e];
    gl[idx] = acc;
}

// ---------------- top-2 + softmax gates + aux accumulation ------------------
__global__ void tg_router_post(const float* __restrict__ gl, float* __restrict__ tw,
                               int* __restrict__ ti, float* __restrict__ acc)
{
    int row = blockIdx.x * blockDim.x + threadIdx.x;
    if (row >= MM) return;
    float v[EE];
#pragma unroll
    for (int e = 0; e < EE; ++e) v[e] = gl[row * EE + e];
    int i1 = 0;
#pragma unroll
    for (int e = 1; e < EE; ++e) if (v[e] > v[i1]) i1 = e;
    int i2 = -1;
#pragma unroll
    for (int e = 0; e < EE; ++e) {
        if (e == i1) continue;
        if (i2 < 0 || v[e] > v[i2]) i2 = e;
    }
    float m  = v[i1];
    float e2 = __expf(v[i2] - m);
    float sw = 1.f + e2;
    tw[row * 2 + 0] = 1.f / sw;
    tw[row * 2 + 1] = e2 / sw;
    ti[row * 2 + 0] = i1;
    ti[row * 2 + 1] = i2;
    float se = 0.f, p[EE];
#pragma unroll
    for (int e = 0; e < EE; ++e) { p[e] = __expf(v[e] - m); se += p[e]; }
#pragma unroll
    for (int e = 0; e < EE; ++e) atomicAdd(&acc[e], p[e] / se);
    atomicAdd(&acc[EE + i1], 1.0f);
}

// ---------------- combine: h += sum_k tw_k * eo[ti_k] -----------------------
__global__ void tg_combine(const float* __restrict__ eo, const float* __restrict__ tw,
                           const int* __restrict__ ti, float* __restrict__ h)
{
    int idx = blockIdx.x * blockDim.x + threadIdx.x;
    if (idx >= MM * DD) return;
    int row = idx / DD;
    int e0 = ti[row * 2 + 0], e1 = ti[row * 2 + 1];
    float o = tw[row * 2 + 0] * eo[(size_t)e0 * (MM * DD) + idx]
            + tw[row * 2 + 1] * eo[(size_t)e1 * (MM * DD) + idx];
    h[idx] += o;
}

__global__ void tg_zero(float* p, int n)
{
    int i = blockIdx.x * blockDim.x + threadIdx.x;
    if (i < n) p[i] = 0.f;
}

__global__ void tg_aux(const float* __restrict__ acc, float* __restrict__ out)
{
    float aux = 0.f;
    for (int l = 0; l < LL; ++l) {
        const float* a = acc + l * 12;
        float s = 0.f;
        for (int e = 0; e < EE; ++e)
            s += (a[e] / (float)MM) * (a[EE + e] / (float)MM);
        aux += 0.01f * (float)EE * s;
    }
    out[0] = aux;
}

// ---------------------------------------------------------------------------
extern "C" void kernel_launch(void* const* d_in, const int* in_sizes, int n_in,
                              void* d_out, int out_size, void* d_ws, size_t ws_size,
                              hipStream_t stream)
{
    (void)in_sizes; (void)n_in; (void)out_size; (void)ws_size;

    const float* x        = (const float*)d_in[0];
    const float* regime   = (const float*)d_in[1];
    const float* film_w1  = (const float*)d_in[2];
    const float* film_b1  = (const float*)d_in[3];
    const float* film_w2  = (const float*)d_in[4];
    const float* film_b2  = (const float*)d_in[5];
    const float* ssm_ng   = (const float*)d_in[6];
    const float* ssm_nb   = (const float*)d_in[7];
    const float* in_w     = (const float*)d_in[8];
    const float* xp_w     = (const float*)d_in[9];
    const float* dt_w     = (const float*)d_in[10];
    const float* dt_b     = (const float*)d_in[11];
    const float* alr      = (const float*)d_in[12];
    const float* aim      = (const float*)d_in[13];
    const float* ssm_d    = (const float*)d_in[14];
    const float* out_w    = (const float*)d_in[15];
    const float* moe_ng   = (const float*)d_in[16];
    const float* moe_nb   = (const float*)d_in[17];
    const float* r_w1     = (const float*)d_in[18];
    const float* r_b1     = (const float*)d_in[19];
    const float* r_w2     = (const float*)d_in[20];
    const float* r_b2     = (const float*)d_in[21];
    const float* e_w1     = (const float*)d_in[22];
    const float* e_b1     = (const float*)d_in[23];
    const float* e_w2     = (const float*)d_in[24];
    const float* e_b2     = (const float*)d_in[25];
    const float* fin_g    = (const float*)d_in[26];
    const float* fin_b    = (const float*)d_in[27];

    // ---- workspace layout: f32 region, then f16 region ----
    float* ws    = (float*)d_ws;
    float* h     = ws;                    // 786432
    float* xz    = h    + 786432;         // 3145728 (x_ssm | z), reused as g1
    float* xdbl  = xz   + 3145728;        // 180224
    float* dtb_  = xdbl + 180224;         // 1572864
    float* y     = dtb_ + 1572864;        // 1572864
    float* gb    = y    + 1572864;        // 3072
    float* rbias = gb   + 3072;           // 1536
    float* gl    = rbias+ 1536;           // 12288
    float* tw    = gl   + 12288;          // 4096
    int*   ti    = (int*)(tw + 4096);     // 4096 ints
    float* macc  = (float*)(ti + 4096);   // 32 (L*2*E = 24 used)
    float* eo    = macc + 32;             // 4718592
    float* g1    = xz;                    // reuse

    _Float16* f16b  = (_Float16*)(eo + 4718592);
    _Float16* xn16  = f16b;               // 786432
    _Float16* xs16  = xn16  + 786432;     // 3145728 (f16 mirror of xz)
    _Float16* y16   = xs16  + 3145728;    // 1572864
    _Float16* eh16  = y16   + 1572864;    // 1048576
    _Float16* wt_in = eh16  + 1048576;    // 589824   (1536 x 384)
    _Float16* wt_xp = wt_in + 589824;     // 67584    (88 x 768)
    _Float16* wt_out= wt_xp + 67584;      // 294912   (384 x 768)
    _Float16* wt_r1 = wt_out+ 294912;     // 147456   (384 x 384)
    _Float16* wt_e1 = wt_r1 + 147456;     // 1179648  (6 x 512 x 384)
    _Float16* wt_e2 = wt_e1 + 1179648;    // 1179648  (6 x 384 x 512)

    float* out = (float*)d_out;

    auto gemm = [&](const _Float16* A, const _Float16* Wt, const float* cb,
                    const float* bbias, const float* Cin, float* Cout,
                    _Float16* CoutH, int M, int N, int K, int lda, int ldc, int act) {
        dim3 grid((N + 127) / 128, M / 64);
        tg_gemm_wmma<<<grid, 256, 0, stream>>>(A, Wt, cb, bbias, Cin, Cout, CoutH,
                                               M, N, K, lda, ldc, act);
    };
    auto w16t = [&](const float* w, _Float16* wt_, int K, int N) {
        tg_w16t<<<(K * N + 255) / 256, 256, 0, stream>>>(w, wt_, K, N);
    };

    // FiLM conditioning
    tg_film<<<BB, 256, 0, stream>>>(regime, film_w1, film_b1, film_w2, film_b2, gb);
    tg_film_apply<<<(MM * DD + 255) / 256, 256, 0, stream>>>(x, gb, h, MM * DD);
    tg_zero<<<1, 32, 0, stream>>>(macc, 32);

    for (int l = 0; l < LL; ++l) {
        const float* in_w_l  = in_w  + (size_t)l * DD * (2 * DIx);
        const float* xp_w_l  = xp_w  + (size_t)l * DIx * NXx;
        const float* dt_w_l  = dt_w  + (size_t)l * DTRx * DIx;
        const float* dt_b_l  = dt_b  + (size_t)l * DIx;
        const float* alr_l   = alr   + (size_t)l * DIx * SS;
        const float* aim_l   = aim   + (size_t)l * DIx * SS;
        const float* d_l     = ssm_d + (size_t)l * DIx;
        const float* out_w_l = out_w + (size_t)l * DIx * DD;
        const float* r_w1_l  = r_w1  + (size_t)l * (DD + RR) * DD;
        const float* r_b1_l  = r_b1  + (size_t)l * DD;
        const float* r_w2_l  = r_w2  + (size_t)l * DD * EE;
        const float* r_b2_l  = r_b2  + (size_t)l * EE;

        // weight caches (f16, transposed) for this layer
        w16t(in_w_l,  wt_in,  DD,  2 * DIx);
        w16t(xp_w_l,  wt_xp,  DIx, NXx);
        w16t(out_w_l, wt_out, DIx, DD);
        w16t(r_w1_l,  wt_r1,  DD,  DD);      // first D rows of r_w1
        for (int e = 0; e < EE; ++e) {
            w16t(e_w1 + ((size_t)l * EE + e) * DD * HH, wt_e1 + (size_t)e * DD * HH, DD, HH);
            w16t(e_w2 + ((size_t)l * EE + e) * HH * DD, wt_e2 + (size_t)e * HH * DD, HH, DD);
        }

        // --- SSM block ---
        tg_ln<<<MM, 128, 0, stream>>>(h, ssm_ng + l * DD, ssm_nb + l * DD, nullptr, xn16);
        gemm(xn16, wt_in, nullptr, nullptr, nullptr, xz, xs16,
             MM, 2 * DIx, DD, DD, 2 * DIx, 0);
        gemm(xs16, wt_xp, nullptr, nullptr, nullptr, xdbl, nullptr,
             MM, NXx, DIx, 2 * DIx, NXx, 0);
        tg_dt<<<(MM * DIx + 255) / 256, 256, 0, stream>>>(xdbl, dt_w_l, dt_b_l, dtb_);
        tg_scan<<<(BB * DIx) / 8, 256, 0, stream>>>(dtb_, xz, xdbl, alr_l, aim_l, y);
        tg_gate<<<(MM * DIx + 255) / 256, 256, 0, stream>>>(xz, d_l, y, y16);
        gemm(y16, wt_out, nullptr, nullptr, h, h, nullptr,
             MM, DD, DIx, DIx, DD, 0);

        // --- MoE block ---
        tg_ln<<<MM, 128, 0, stream>>>(h, moe_ng + l * DD, moe_nb + l * DD, nullptr, xn16);
        tg_rbias<<<BB, DD, 0, stream>>>(regime, r_w1_l, r_b1_l, rbias);
        gemm(xn16, wt_r1, nullptr, rbias, nullptr, g1, nullptr,
             MM, DD, DD, DD, DD, 1);
        tg_gl<<<(MM * EE + 255) / 256, 256, 0, stream>>>(g1, r_w2_l, r_b2_l, gl);
        tg_router_post<<<(MM + 255) / 256, 256, 0, stream>>>(gl, tw, ti, macc + l * 12);

        for (int e = 0; e < EE; ++e) {
            const float* b1e = e_b1 + ((size_t)l * EE + e) * HH;
            const float* b2e = e_b2 + ((size_t)l * EE + e) * DD;
            gemm(xn16, wt_e1 + (size_t)e * DD * HH, b1e, nullptr, nullptr, nullptr,
                 eh16, MM, HH, DD, DD, HH, 1);
            gemm(eh16, wt_e2 + (size_t)e * HH * DD, b2e, nullptr, nullptr,
                 eo + (size_t)e * MM * DD, nullptr, MM, DD, HH, HH, DD, 0);
        }
        tg_combine<<<(MM * DD + 255) / 256, 256, 0, stream>>>(eo, tw, ti, h);
    }

    // final LN -> out, aux scalar
    tg_ln<<<MM, 128, 0, stream>>>(h, fin_g, fin_b, out, nullptr);
    tg_aux<<<1, 1, 0, stream>>>(macc, out + (size_t)MM * DD);
}
